// GELU217_23648089932089
// MI455X (gfx1250) — matrix operand order repaired and checked by reading, not compiled, not run on hardware
//
#include <hip/hip_runtime.h>

typedef __bf16 bf16_t;
typedef __attribute__((ext_vector_type(16))) __bf16 v16bf;
typedef __attribute__((ext_vector_type(8)))  __bf16 v8bf;
typedef __attribute__((ext_vector_type(8)))  float  v8f;

static constexpr int D_DIM = 512;
static constexpr int T_DIM = 2048;
static constexpr int B_DIM = 8;

__device__ __forceinline__ float softplus_f(float v) { return log1pf(expf(v)); }
__device__ __forceinline__ float gelu_tanh_f(float x) {
    const float k = 0.7978845608028654f;
    return 0.5f * x * (1.0f + tanhf(k * (x + 0.044715f * x * x * x)));
}

// ---- CDNA5 async global->LDS helpers (ASYNCcnt-tracked) --------------------
__device__ __forceinline__ void wait_async0() {
    asm volatile("s_wait_asynccnt 0" ::: "memory");
}
__device__ __forceinline__ void wait_ds0() {
    asm volatile("s_wait_dscnt 0" ::: "memory");
}
// One b128 async copy per lane: LDS[lds_addr] = MEM[saddr + voff]
__device__ __forceinline__ void async_b128(unsigned lds_addr, unsigned voff,
                                           const void* saddr) {
    asm volatile("global_load_async_to_lds_b128 %0, %1, %2 offset:0"
                 :: "v"(lds_addr), "v"(voff), "s"(saddr)
                 : "memory");
}
// Stage one quarter (per 128-thread block) of a 16x512 bf16 tile: 8 async
// b128 ops per thread, adjacent threads -> adjacent 16B chunks (coalesced).
__device__ __forceinline__ void stage_async_quarter(const bf16_t* xn,
                                                    size_t elem_base,
                                                    unsigned lds_base, int tid) {
    const unsigned g0 = (unsigned)(elem_base * 2) + (unsigned)tid * 16u;
    const unsigned l0 = lds_base + (unsigned)tid * 16u;
    #pragma unroll
    for (int t = 0; t < 8; ++t)
        async_b128(l0 + (unsigned)t * 2048u, g0 + (unsigned)t * 2048u, xn);
}

// ---------------------------------------------------------------------------
// Kernel A: per-row elementwise path + row reductions.
// One wave32 per row (D=512 -> 16 contiguous elements per lane).
// Writes out * gate_hist * gate_cos to d_out, normalized x (bf16) to ws.
// ---------------------------------------------------------------------------
__global__ __launch_bounds__(256) void gelu_gate_row_kernel(
    const float* __restrict__ x, const float* __restrict__ ema_mean,
    const float* __restrict__ ema_sq, const float* __restrict__ ema_out,
    const float* __restrict__ p_log_tau, const float* __restrict__ p_lbu,
    const float* __restrict__ p_lbd, const float* __restrict__ p_lg,
    float* __restrict__ out, bf16_t* __restrict__ xn)
{
    const int wave = threadIdx.x >> 5;
    const int lane = threadIdx.x & 31;
    const int row  = blockIdx.x * 8 + wave;            // [0, B*T)

    const float tau     = expf(p_log_tau[0]);
    const float beta_up = softplus_f(p_lbu[0]);
    const float beta_dn = softplus_f(p_lbd[0]);
    const float gamma   = softplus_f(p_lg[0]);

    const float* xr = x + (size_t)row * D_DIM;
    const int c0 = lane * 16;

    float xv[16], pv[16];
    float sx = 0.f, so = 0.f, sd = 0.f, se = 0.f;
    #pragma unroll
    for (int i = 0; i < 16; ++i) {
        const int c = c0 + i;
        const float xi = xr[c];
        const float m  = ema_mean[c];
        const float s2 = ema_sq[c];
        const float eo = ema_out[c];
        const float o  = gelu_tanh_f(xi);
        const float var = fmaxf(s2 - m * m, 1e-4f);
        const float z   = (xi - m) / (sqrtf(var) + 1e-5f);
        const float up  = beta_up * fmaxf(tanhf(gamma * z), 0.f);
        const float dn  = beta_dn * fmaxf(tanhf(-gamma * z), 0.f);
        const float gh  = fminf(fmaxf(1.f + up - dn, 0.05f), 8.f);
        xv[i] = xi;
        pv[i] = o * gh;
        sx += xi * xi; so += o * o; sd += o * eo; se += eo * eo;
    }
    #pragma unroll
    for (int m = 16; m >= 1; m >>= 1) {
        sx += __shfl_xor(sx, m, 32);
        so += __shfl_xor(so, m, 32);
        sd += __shfl_xor(sd, m, 32);
        se += __shfl_xor(se, m, 32);
    }
    const float nx = fmaxf(sqrtf(sx), 1e-12f);
    const float no = fmaxf(sqrtf(so), 1e-12f);
    const float ne = fmaxf(sqrtf(se), 1e-12f);
    const float cs = fminf(fmaxf(sd / (no * ne), -1.f), 1.f);
    const float gcos = expf(-tau * cs);
    const float inx  = 1.f / nx;

    float*  orow  = out + (size_t)row * D_DIM;
    bf16_t* xnrow = xn  + (size_t)row * D_DIM;
    #pragma unroll
    for (int i = 0; i < 16; ++i) {
        const int c = c0 + i;
        orow[c]  = pv[i] * gcos;
        xnrow[c] = (bf16_t)(xv[i] * inx);
    }
}

// ---------------------------------------------------------------------------
// Kernel B: Gram matrix tiles via V_WMMA_F32_16X16X32_BF16.
// 4 waves per block; wave w owns i-tile 4*blockTile+w with its A-fragments
// resident in 128 VGPRs. All 4 waves share one async double-buffered B tile
// stream (GLOBAL_LOAD_ASYNC_TO_LDS_B128). The K reduction runs as TWO
// independent chains (even/odd chunks, separate accumulators and B register
// pairs) so each chain's LDS loads overlap the other chain's WMMA.
// ---------------------------------------------------------------------------
__global__ __launch_bounds__(128) void gram_nn_gate_kernel(
    const bf16_t* __restrict__ xn, float* __restrict__ out,
    const float* __restrict__ p_lsig, const float* __restrict__ p_lw)
{
    __shared__ __bf16 lB[2][16 * 512];         // 2 x 16 KB double buffer
    __shared__ float  sims[4][16];
    __shared__ float  gate_s[4][16];

    const int tid   = threadIdx.x;
    const int wave  = tid >> 5;
    const int lane  = tid & 31;
    const int btile = blockIdx.x & 31;         // 64-row super-tile index
    const int b     = blockIdx.x >> 5;
    const int i0    = (btile * 4 + wave) * 16; // this wave's 16-row i-tile
    const size_t base = (size_t)b * T_DIM * D_DIM;

    const int half = lane >> 4;
    const int mn   = lane & 15;                // A: row M; B/C: col N
    const int kbA  = half ? 16 : 0;            // A-frag K base (contiguous k = 16*half + s)
    const int kbB  = half ? 16 : 0;            // B-frag K base (contiguous k = 16*half + s)

    // Gather this wave's 16 A-fragments straight from global into registers
    // (one-time 16 KB per wave; two b128 loads per K-chunk per lane).
    v16bf afrag[16];
    {
        const bf16_t* arow = xn + base + (size_t)(i0 + mn) * D_DIM;
        #pragma unroll
        for (int kc = 0; kc < 16; ++kc) {
            const int kk = kc * 32;
            v8bf a0 = *reinterpret_cast<const v8bf*>(arow + kk + kbA);
            v8bf a1 = *reinterpret_cast<const v8bf*>(arow + kk + 8 + kbA);
            #pragma unroll
            for (int e = 0; e < 8; ++e) { afrag[kc][e] = a0[e]; afrag[kc][8 + e] = a1[e]; }
        }
    }

    const unsigned ldsB0 = (unsigned)(uintptr_t)(&lB[0][0]);
    const unsigned ldsB1 = (unsigned)(uintptr_t)(&lB[1][0]);

    float rmax[8];
    #pragma unroll
    for (int r = 0; r < 8; ++r) rmax[r] = -2.0f;

    // Prologue: all 4 waves cooperatively async-stage tile j=0 into buffer 0.
    stage_async_quarter(xn, base, ldsB0, tid);
    wait_async0();
    __syncthreads();

    for (int j = 0; j < 128; ++j) {
        const int p = j & 1;
        if (j + 1 < 128)                        // prefetch next tile -> buf 1-p
            stage_async_quarter(xn, base + (size_t)(j + 1) * 16 * D_DIM,
                                p ? ldsB0 : ldsB1, tid);

        const __bf16* curB = p ? &lB[1][0] : &lB[0][0];
        const __bf16* rowB = &curB[mn * 512 + kbB];

        // Two independent chains: even chunks -> acc0, odd chunks -> acc1.
        v8f acc0 = {0.f, 0.f, 0.f, 0.f, 0.f, 0.f, 0.f, 0.f};
        v8f acc1 = acc0;
        // Chain prologues: both chains' B pairs live simultaneously.
        v8bf e0 = *reinterpret_cast<const v8bf*>(rowB + 0);
        v8bf e1 = *reinterpret_cast<const v8bf*>(rowB + 8);
        v8bf o0 = *reinterpret_cast<const v8bf*>(rowB + 32);
        v8bf o1 = *reinterpret_cast<const v8bf*>(rowB + 40);
        #pragma unroll
        for (int kc = 0; kc < 16; kc += 2) {
            v16bf bve;
            #pragma unroll
            for (int e = 0; e < 8; ++e) { bve[e] = e0[e]; bve[8 + e] = e1[e]; }
            if (kc + 2 < 16) {                 // prefetch next even chunk
                e0 = *reinterpret_cast<const v8bf*>(rowB + (kc + 2) * 32);
                e1 = *reinterpret_cast<const v8bf*>(rowB + (kc + 2) * 32 + 8);
            }
            acc0 = __builtin_amdgcn_wmma_f32_16x16x32_bf16(
                false, afrag[kc], false, bve, (short)0, acc0, false, false);

            v16bf bvo;
            #pragma unroll
            for (int e = 0; e < 8; ++e) { bvo[e] = o0[e]; bvo[8 + e] = o1[e]; }
            if (kc + 3 < 16) {                 // prefetch next odd chunk
                o0 = *reinterpret_cast<const v8bf*>(rowB + (kc + 3) * 32);
                o1 = *reinterpret_cast<const v8bf*>(rowB + (kc + 3) * 32 + 8);
            }
            acc1 = __builtin_amdgcn_wmma_f32_16x16x32_bf16(
                false, afrag[kc + 1], false, bvo, (short)0, acc1, false, false);
        }

        // Combine chains, mask diagonal, accumulate per-lane running max.
        const int jc = j * 16 + mn;            // global column
        #pragma unroll
        for (int r = 0; r < 8; ++r) {
            const int gr = i0 + r + (half ? 8 : 0);   // global row (C layout)
            const float s = acc0[r] + acc1[r];
            const float v = (gr == jc) ? -2.0f : s;
            rmax[r] = fmaxf(rmax[r], v);
        }

        wait_ds0();                            // this wave's reads drained
        wait_async0();                         // this wave's prefetch landed
        __syncthreads();                       // publish prefetch block-wide
    }

    // Reduce across the 16 lanes of each half (masks < 16 keep halves apart).
    #pragma unroll
    for (int m = 1; m < 16; m <<= 1) {
        #pragma unroll
        for (int r = 0; r < 8; ++r)
            rmax[r] = fmaxf(rmax[r], __shfl_xor(rmax[r], m, 32));
    }
    if (mn == 0) {
        #pragma unroll
        for (int r = 0; r < 8; ++r) sims[wave][r + (half ? 8 : 0)] = rmax[r];
    }
    __syncthreads();

    if (tid < 64) {
        const float v   = sims[tid >> 4][tid & 15];   // nn_sim for block row
        const float sig = softplus_f(p_lsig[0]);
        const float w   = softplus_f(p_lw[0]);
        gate_s[tid >> 4][tid & 15] = 1.0f + w * tanhf(sig * 0.5f * (1.0f - v));
    }
    __syncthreads();

    // Scale this block's 64 output rows in place (64 x 128 float4).
    float4* orow = reinterpret_cast<float4*>(out + base + (size_t)(btile * 64) * D_DIM);
    for (int idx = tid; idx < 64 * 128; idx += 128) {
        const int r = idx >> 7;
        const float g = gate_s[r >> 4][r & 15];
        float4 v = orow[idx];
        v.x *= g; v.y *= g; v.z *= g; v.w *= g;
        orow[idx] = v;
    }
}

extern "C" void kernel_launch(void* const* d_in, const int* in_sizes, int n_in,
                              void* d_out, int out_size, void* d_ws, size_t ws_size,
                              hipStream_t stream) {
    const float* x        = (const float*)d_in[0];
    const float* ema_mean = (const float*)d_in[1];
    const float* ema_sq   = (const float*)d_in[2];
    const float* ema_out  = (const float*)d_in[3];
    const float* log_tau  = (const float*)d_in[4];
    const float* lbu      = (const float*)d_in[5];
    const float* lbd      = (const float*)d_in[6];
    const float* lg       = (const float*)d_in[7];
    const float* lsig     = (const float*)d_in[8];
    const float* lw       = (const float*)d_in[9];
    float* out = (float*)d_out;
    bf16_t* xn = (bf16_t*)d_ws;   // B*T*D bf16 = 16 MB of scratch

    // Kernel A: one wave per row, 8 rows per 256-thread block.
    gelu_gate_row_kernel<<<(B_DIM * T_DIM) / 8, 256, 0, stream>>>(
        x, ema_mean, ema_sq, ema_out, log_tau, lbu, lbd, lg, out, xn);

    // Kernel B: 4 waves per block, each owning a 16-row i-tile; 64 rows/block.
    gram_nn_gate_kernel<<<B_DIM * (T_DIM / 64), 128, 0, stream>>>(
        xn, out, lsig, lw);
}
